// DotProductAttention_32212254720701
// MI455X (gfx1250) — matrix operand order repaired and checked by reading, not compiled, not run on hardware
//
#include <hip/hip_runtime.h>

typedef __attribute__((ext_vector_type(16))) _Float16 v16h;
typedef __attribute__((ext_vector_type(8)))  _Float16 v8h;
typedef __attribute__((ext_vector_type(8)))  float    v8f;

namespace {
constexpr int   Dh    = 128;    // head dim
constexpr int   LK    = 2048;   // keys
constexpr int   LQ    = 2048;   // queries
constexpr int   KT    = 64;     // keys per tile
constexpr int   KSTR  = Dh + 8; // LDS K row stride in halves (272B, 16B aligned)
constexpr int   VSTR  = KT + 8; // LDS V^T row stride in halves (144B, 16B aligned)
constexpr float NEGBIG = -1.0e9f;
constexpr float LOG2E  = 1.4426950408889634f;

constexpr int SZ_K  = KT * KSTR * 2;   // 17408 B
constexpr int SZ_VT = Dh * VSTR * 2;   // 18432 B
constexpr int SZ_B  = LK * 4;          //  8192 B
}

__global__ __launch_bounds__(128, 1)
void fa_fwd_kernel(const float* __restrict__ Q, const float* __restrict__ K,
                   const float* __restrict__ V, const float* __restrict__ Msk,
                   float* __restrict__ O)
{
    // 44032 B total; output staging (32 KB per pass) aliases the K/V tile region.
    __shared__ __align__(16) unsigned char smem[SZ_K + SZ_VT + SZ_B];
    _Float16* ldsK    = (_Float16*)smem;
    _Float16* ldsVT   = (_Float16*)(smem + SZ_K);
    float*    biasLds = (float*)(smem + SZ_K + SZ_VT);
    float*    oLds    = (float*)smem;                   // epilogue alias

    const int tid  = threadIdx.x;
    const int lane = tid & 31;
    const int wv   = tid >> 5;
    const int l16  = lane & 15;
    const int hi   = lane >> 4;                 // 0: lanes 0-15, 1: lanes 16-31

    const int bid   = blockIdx.x;
    const int b     = bid >> 4;                 // / (LQ/128)
    const int qbase = (bid & 15) * 128;         // 128 q rows per block

    const float* Qb = Q + (size_t)b * LQ * Dh;
    const float* Kb = K + (size_t)b * LK * Dh;
    const float* Vb = V + (size_t)b * LK * Dh;
    float*       Ob = O + (size_t)b * LQ * Dh;

    const float kscale = LOG2E * 0.08838834764831845f;  // log2(e)/sqrt(128)

    // ---- precompute masked bias into LDS (exp2 domain) ----
    for (int i = tid; i < LK; i += 128)
        biasLds[i] = Msk[i] * (NEGBIG * LOG2E);

    // ---- load Q B-fragments: two 16-column blocks per wave (32 q rows) ----
    v16h Qf[2][4];
    #pragma unroll
    for (int n = 0; n < 2; ++n) {
        const int qrow = qbase + wv * 32 + 16 * n + l16;
        const float* qp = Qb + (size_t)qrow * Dh + hi * 8;
        #pragma unroll
        for (int c = 0; c < 4; ++c) {
            const float4* p = (const float4*)(qp + 32 * c);
            #pragma unroll
            for (int j = 0; j < 4; ++j) {
                float4 x = p[j + 2 * (j >> 1)];
                Qf[n][c][4*j+0] = (_Float16)x.x;  Qf[n][c][4*j+1] = (_Float16)x.y;
                Qf[n][c][4*j+2] = (_Float16)x.z;  Qf[n][c][4*j+3] = (_Float16)x.w;
            }
        }
    }

    // ---- accumulators ----
    v8f Oacc[2][8];
    #pragma unroll
    for (int n = 0; n < 2; ++n)
        #pragma unroll
        for (int f = 0; f < 8; ++f)
            #pragma unroll
            for (int r = 0; r < 8; ++r) Oacc[n][f][r] = 0.0f;
    float mrun[2] = {-3.0e38f, -3.0e38f};
    float lrun[2] = {0.0f, 0.0f};

    for (int kb = 0; kb < LK; kb += KT) {
        // ---- stage K tile -> LDS f16 row-major (64 keys x 128 d) ----
        {
            const int key = tid >> 1, ch = tid & 1;    // key row, 64-d half
            const float* kp = Kb + (size_t)(kb + key) * Dh + 64 * ch;
            if (kb + KT < LK) {
                __builtin_prefetch(kp + KT * Dh, 0, 1);
                __builtin_prefetch(Vb + (size_t)(kb + KT + (tid >> 1)) * Dh + 64 * (tid & 1), 0, 1);
            }
            #pragma unroll
            for (int j = 0; j < 8; ++j) {
                float4 a  = ((const float4*)kp)[2*j];
                float4 bq = ((const float4*)kp)[2*j+1];
                v8h hv;
                hv[0]=(_Float16)a.x;  hv[1]=(_Float16)a.y;  hv[2]=(_Float16)a.z;  hv[3]=(_Float16)a.w;
                hv[4]=(_Float16)bq.x; hv[5]=(_Float16)bq.y; hv[6]=(_Float16)bq.z; hv[7]=(_Float16)bq.w;
                *(v8h*)&ldsK[key * KSTR + 64 * ch + 8 * j] = hv;
            }
        }
        // ---- stage V tile transposed -> LDS f16 [d][key] ----
        {
            const int g = tid & 15, q8 = tid >> 4;   // d-group (8 d's), key-octet (8 keys)
            float vv[8][8];
            #pragma unroll
            for (int i = 0; i < 8; ++i) {
                const float* vp = Vb + (size_t)(kb + 8*q8 + i) * Dh + 8 * g;
                float4 a  = ((const float4*)vp)[0];
                float4 c2 = ((const float4*)vp)[1];
                vv[i][0]=a.x;  vv[i][1]=a.y;  vv[i][2]=a.z;  vv[i][3]=a.w;
                vv[i][4]=c2.x; vv[i][5]=c2.y; vv[i][6]=c2.z; vv[i][7]=c2.w;
            }
            #pragma unroll
            for (int dd = 0; dd < 8; ++dd) {
                v8h pk;
                #pragma unroll
                for (int i = 0; i < 8; ++i) pk[i] = (_Float16)vv[i][dd];
                *(v8h*)&ldsVT[(8*g + dd) * VSTR + 8*q8] = pk;
            }
        }
        __syncthreads();

        // ---- S^T = K_tile x Q^T : each K A-fragment feeds BOTH q-column blocks ----
        v8f s[2][4];
        #pragma unroll
        for (int n = 0; n < 2; ++n)
            #pragma unroll
            for (int g = 0; g < 4; ++g)
                #pragma unroll
                for (int r = 0; r < 8; ++r) s[n][g][r] = 0.0f;
        #pragma unroll
        for (int g = 0; g < 4; ++g) {
            const _Float16* krow = &ldsK[(16*g + l16) * KSTR];
            #pragma unroll
            for (int c = 0; c < 4; ++c) {
                const int dlo = 32 * c + 8 * hi;
                v8h lo = *(const v8h*)(krow + dlo);
                v8h hh = *(const v8h*)(krow + dlo + 16);
                v16h a;
                #pragma unroll
                for (int i = 0; i < 8; ++i) { a[i] = lo[i]; a[8+i] = hh[i]; }
                #pragma unroll
                for (int n = 0; n < 2; ++n)
                    s[n][g] = __builtin_amdgcn_wmma_f32_16x16x32_f16(
                                false, a, false, Qf[n][c], (short)0, s[n][g], false, false);
            }
        }

        // ---- scale + key-padding mask bias (exp2 domain) ----
        #pragma unroll
        for (int g = 0; g < 4; ++g) {
            const float4* bp = (const float4*)&biasLds[kb + 16*g + 8*hi];
            float4 ba = bp[0], bb = bp[1];
            float bias[8] = {ba.x,ba.y,ba.z,ba.w, bb.x,bb.y,bb.z,bb.w};
            #pragma unroll
            for (int n = 0; n < 2; ++n)
                #pragma unroll
                for (int r = 0; r < 8; ++r)
                    s[n][g][r] = fmaf(s[n][g][r], kscale, bias[r]);
        }

        // ---- online softmax (per q-column block) ----
        v16h Pb[2][2];
        #pragma unroll
        for (int n = 0; n < 2; ++n) {
            float tmax = s[n][0][0];
            #pragma unroll
            for (int g = 0; g < 4; ++g)
                #pragma unroll
                for (int r = 0; r < 8; ++r) tmax = fmaxf(tmax, s[n][g][r]);
            tmax = fmaxf(tmax, __shfl_xor(tmax, 16, 32));
            const float mnew  = fmaxf(mrun[n], tmax);
            const float alpha = exp2f(mrun[n] - mnew);
            float psum = 0.0f;
            #pragma unroll
            for (int g = 0; g < 4; ++g)
                #pragma unroll
                for (int r = 0; r < 8; ++r) { s[n][g][r] = exp2f(s[n][g][r] - mnew); psum += s[n][g][r]; }
            psum += __shfl_xor(psum, 16, 32);
            lrun[n] = lrun[n] * alpha + psum;
            mrun[n] = mnew;
            #pragma unroll
            for (int f = 0; f < 8; ++f)
                #pragma unroll
                for (int r = 0; r < 8; ++r) Oacc[n][f][r] *= alpha;

            // build P^T B-fragments from S^T C-layout (one shfl_xor(16) per reg)
            #pragma unroll
            for (int p = 0; p < 2; ++p) {
                #pragma unroll
                for (int r = 0; r < 8; ++r) {
                    float x0 = __shfl_xor(s[n][2*p][r],   16, 32);
                    float x1 = __shfl_xor(s[n][2*p+1][r], 16, 32);
                    float loe = hi ? x1 : s[n][2*p][r];     // k = 32p+0..7   / 32p+16..23
                    float hie = hi ? s[n][2*p+1][r] : x0;   // k = 32p+8..15  / 32p+24..31
                    Pb[n][p][r]     = (_Float16)loe;
                    Pb[n][p][8 + r] = (_Float16)hie;
                }
            }
        }

        // ---- O^T += V^T x P^T : each V A-fragment feeds BOTH q-column blocks ----
        #pragma unroll
        for (int f = 0; f < 8; ++f) {
            const _Float16* vrow = &ldsVT[(16*f + l16) * VSTR];
            #pragma unroll
            for (int p = 0; p < 2; ++p) {
                const int k0 = 32 * p + 16 * hi;
                v8h lo = *(const v8h*)(vrow + k0);
                v8h hh = *(const v8h*)(vrow + k0 + 8);
                v16h a;
                #pragma unroll
                for (int i = 0; i < 8; ++i) { a[i] = lo[i]; a[8+i] = hh[i]; }
                #pragma unroll
                for (int n = 0; n < 2; ++n)
                    Oacc[n][f] = __builtin_amdgcn_wmma_f32_16x16x32_f16(
                                   false, a, false, Pb[n][p], (short)0, Oacc[n][f], false, false);
            }
        }
        __syncthreads();
    }

    // ---- epilogue: normalize, transpose through LDS, coalesced stores (2 passes) ----
    #pragma unroll
    for (int n = 0; n < 2; ++n) {
        const float invl = (lrun[n] > 0.0f) ? (1.0f / lrun[n]) : 0.0f;
        #pragma unroll
        for (int f = 0; f < 8; ++f)
            #pragma unroll
            for (int r = 0; r < 8; ++r)
                oLds[(wv*16 + l16) * Dh + 16*f + r + 8*hi] = Oacc[n][f][r] * invl;
        __syncthreads();
        {
            // oLds row r (0..63) -> global q row = qbase + (r/16)*32 + 16*n + (r%16)
            const int row  = tid >> 1;
            const int half = tid & 1;
            const int grow = qbase + (row >> 4) * 32 + 16 * n + (row & 15);
            float4* dst = (float4*)(Ob + (size_t)grow * Dh + 64 * half);
            const float4* s4 = (const float4*)(oLds + row * Dh + 64 * half);
            #pragma unroll
            for (int j = 0; j < 16; ++j)
                dst[j] = s4[j];
        }
        __syncthreads();
    }
}

extern "C" void kernel_launch(void* const* d_in, const int* in_sizes, int n_in,
                              void* d_out, int out_size, void* d_ws, size_t ws_size,
                              hipStream_t stream) {
    const float* Q = (const float*)d_in[0];
    const float* K = (const float*)d_in[1];
    const float* V = (const float*)d_in[2];
    const float* M = (const float*)d_in[3];
    float* O = (float*)d_out;
    (void)in_sizes; (void)n_in; (void)out_size; (void)d_ws; (void)ws_size;
    dim3 grid(16 * (LQ / 128));   // 256 blocks: batch x 128-row q-tile
    fa_fwd_kernel<<<grid, 128, 0, stream>>>(Q, K, V, M, O);
}
